// NukeGATPredictor_55731495633464
// MI455X (gfx1250) — matrix operand order repaired and checked by reading, not compiled, not run on hardware
//
#include <hip/hip_runtime.h>
#include <math.h>

typedef __attribute__((ext_vector_type(16))) __bf16 v16bf;
typedef __attribute__((ext_vector_type(8)))  __bf16 v8bf;
typedef __attribute__((ext_vector_type(8)))  float  v8f;

#define DEVFN __device__ __forceinline__

// ---------------- problem constants ----------------
constexpr int N_NODES = 10000;
constexpr int N_EDGES = 160000;
constexpr int ET      = N_EDGES + N_NODES;   // edges + self loops = 170000
constexpr int F_IN    = 64;
constexpr int CH      = 128;                 // per-head channels
constexpr int NH      = 8;                   // heads
constexpr int HC      = 1024;                // NH*CH
constexpr int NG      = 64;                  // graphs
constexpr int NCLS    = 128;

// =====================================================================
// fp32 -> bf16 streaming conversion (vectorized by 4)
// =====================================================================
__global__ __launch_bounds__(256)
void cvt_f32_to_bf16(const float* __restrict__ in, __bf16* __restrict__ out, int n)
{
  const int i = (blockIdx.x * 256 + threadIdx.x) * 4;
  if (i >= n) return;
  const float4 v = *(const float4*)(in + i);
  out[i + 0] = (__bf16)v.x;
  out[i + 1] = (__bf16)v.y;
  out[i + 2] = (__bf16)v.z;
  out[i + 3] = (__bf16)v.w;
}

// =====================================================================
// WMMA bf16 GEMM:  OUT[M,Nout] = X[M,K] @ W[Nout,K]^T (+ bias)
// X, W are pre-converted bf16.  block = 128 threads (4 waves); each wave
// computes a 16(M) x 64(N) strip with 4 f32 accumulators, A fragment
// reused across the 4 WMMAs.  M%16==0, Nout%64==0, K%32==0 here.
// =====================================================================
DEVFN v16bf load16bf(const __bf16* p0, const __bf16* p1) {
  union { v16bf v; v8bf h[2]; } u;
  u.h[0] = *(const v8bf*)p0;   // 16B load
  u.h[1] = *(const v8bf*)p1;   // 16B load
  return u.v;
}

__global__ __launch_bounds__(128)
void gemm_bf16_wmma(const __bf16* __restrict__ X, const __bf16* __restrict__ W,
                    const float* __restrict__ bias, float* __restrict__ OUT,
                    int M, int Nout, int K)
{
  const int lane = threadIdx.x & 31;
  const int wave = threadIdx.x >> 5;
  const int m0 = (blockIdx.y * 4 + wave) << 4;
  const int n0 = blockIdx.x << 6;
  if (m0 >= M) return;                       // wave-uniform exit (EXEC all-1 for WMMA)

  const int ln = lane & 15;                  // row (A) / col (B,D)
  const int kh = lane >> 4;                  // K-half selector

  v8f acc[4] = {v8f{}, v8f{}, v8f{}, v8f{}};

  const __bf16* arow = X + (size_t)(m0 + ln) * K + (kh << 3);

  for (int k = 0; k < K; k += 32) {
    // A fragment (16x32 bf16): elems 0..7 = K k..k+7 (lanes>=16: +8), 8..15 = +16
    const __bf16* ap = arow + k;
    if (k + 32 < K) __builtin_prefetch(ap + 32, 0, 1);
    const v16bf afr = load16bf(ap, ap + 16);

#pragma unroll
    for (int t = 0; t < 4; ++t) {
      // B fragment (32x16 bf16): lane=col n, elems 0..7 = K k..k+7 (lanes>=16: +8), 8..15 = +16
      const __bf16* brow = W + (size_t)(n0 + t * 16 + ln) * K + (kh << 3) + k;
      const v16bf bfr = load16bf(brow, brow + 16);
      acc[t] = __builtin_amdgcn_wmma_f32_16x16x32_bf16(
          false, afr, false, bfr, (short)0, acc[t], false, false);
    }
  }

  // D layout: VGPR r -> row m0 + kh*8 + r, col n0 + t*16 + (lane&15)
#pragma unroll
  for (int t = 0; t < 4; ++t) {
    const int col = n0 + t * 16 + ln;
    const float bv = bias ? bias[col] : 0.0f;
    float* orow = OUT + (size_t)(m0 + (kh << 3)) * Nout + col;
#pragma unroll
    for (int r = 0; r < 8; ++r)
      orow[(size_t)r * Nout] = acc[t][r] + bv;
  }
}

// =====================================================================
// Fused (optional bias + beta*res) + LayerNorm + exact GELU, one row/block
// =====================================================================
__global__ __launch_bounds__(256)
void fused_res_ln_gelu(const float* __restrict__ in, const float* __restrict__ bias,
                       const float* __restrict__ res, const float* __restrict__ beta_p,
                       const float* __restrict__ g, const float* __restrict__ be,
                       float* __restrict__ out, int D)
{
  __shared__ float buf[HC];
  __shared__ float red[256];
  const int row = blockIdx.x;
  const size_t base = (size_t)row * D;
  const float beta = res ? beta_p[0] : 0.0f;

  float s = 0.0f;
  for (int c = threadIdx.x; c < D; c += 256) {
    float v = in[base + c];
    if (bias) v += bias[c];
    if (res)  v += beta * res[base + c];
    buf[c] = v;
    s += v;
  }
  red[threadIdx.x] = s; __syncthreads();
  for (int o = 128; o; o >>= 1) {
    if (threadIdx.x < o) red[threadIdx.x] += red[threadIdx.x + o];
    __syncthreads();
  }
  const float mean = red[0] / (float)D;
  __syncthreads();

  float s2 = 0.0f;
  for (int c = threadIdx.x; c < D; c += 256) {
    const float dv = buf[c] - mean;
    s2 += dv * dv;
  }
  red[threadIdx.x] = s2; __syncthreads();
  for (int o = 128; o; o >>= 1) {
    if (threadIdx.x < o) red[threadIdx.x] += red[threadIdx.x + o];
    __syncthreads();
  }
  const float inv = rsqrtf(red[0] / (float)D + 1e-5f);

  for (int c = threadIdx.x; c < D; c += 256) {
    const float y = (buf[c] - mean) * inv * g[c] + be[c];
    out[base + c] = 0.5f * y * (1.0f + erff(y * 0.70710678118f));  // exact GELU
  }
}

// =====================================================================
// Edge kernels (GATv2 attention).  One wave per (edge, head).
// =====================================================================
DEVFN void edge_endpoints(const int* __restrict__ ei, int e, int& s, int& d) {
  if (e < N_EDGES) { s = ei[e]; d = ei[N_EDGES + e]; }
  else             { s = d = e - N_EDGES; }              // self loops
}

__global__ __launch_bounds__(256)
void edge_scores(const float* __restrict__ XL, const float* __restrict__ XR,
                 const float* __restrict__ att, const int* __restrict__ ei,
                 float* __restrict__ SC)
{
  const int gw = blockIdx.x * 8 + (threadIdx.x >> 5);   // global wave id
  const int lane = threadIdx.x & 31;
  if (gw >= ET * NH) return;
  const int h = gw & 7;
  const int e = gw >> 3;
  int s, d; edge_endpoints(ei, e, s, d);

  const float* pl = XL + (size_t)s * HC + h * CH + lane * 4;
  const float* pr = XR + (size_t)d * HC + h * CH + lane * 4;
  const float* pa = att + h * CH + lane * 4;
  float acc = 0.0f;
#pragma unroll
  for (int j = 0; j < 4; ++j) {
    float v = pl[j] + pr[j];
    v = v > 0.0f ? v : 0.2f * v;                        // leaky_relu(0.2)
    acc += v * pa[j];
  }
#pragma unroll
  for (int off = 16; off; off >>= 1) acc += __shfl_xor(acc, off, 32);
  if (lane == 0) SC[(size_t)e * NH + h] = acc;
}

DEVFN void atomicMaxFloat(float* addr, float v) {
  if (v >= 0.0f) atomicMax((int*)addr, __float_as_int(v));
  else           atomicMin((unsigned int*)addr, __float_as_uint(v));
}

__global__ __launch_bounds__(256)
void edge_seg_max(const float* __restrict__ SC, const int* __restrict__ ei,
                  float* __restrict__ MAXB)
{
  const int idx = blockIdx.x * 256 + threadIdx.x;
  if (idx >= ET * NH) return;
  const int h = idx & 7, e = idx >> 3;
  int s, d; edge_endpoints(ei, e, s, d); (void)s;
  atomicMaxFloat(&MAXB[d * NH + h], SC[idx]);
}

__global__ __launch_bounds__(256)
void edge_exp_sum(float* __restrict__ SC, const int* __restrict__ ei,
                  const float* __restrict__ MAXB, float* __restrict__ SUMB)
{
  const int idx = blockIdx.x * 256 + threadIdx.x;
  if (idx >= ET * NH) return;
  const int h = idx & 7, e = idx >> 3;
  int s, d; edge_endpoints(ei, e, s, d); (void)s;
  const float ex = expf(SC[idx] - MAXB[d * NH + h]);
  SC[idx] = ex;
  atomicAdd(&SUMB[d * NH + h], ex);
}

__global__ __launch_bounds__(256)
void edge_scatter(const float* __restrict__ SC, const float* __restrict__ SUMB,
                  const float* __restrict__ XL, const int* __restrict__ ei,
                  float* __restrict__ AGG)
{
  const int gw = blockIdx.x * 8 + (threadIdx.x >> 5);
  const int lane = threadIdx.x & 31;
  if (gw >= ET * NH) return;
  const int h = gw & 7;
  const int e = gw >> 3;
  int s, d; edge_endpoints(ei, e, s, d);
  const float alpha = SC[(size_t)e * NH + h] / SUMB[d * NH + h];
  const float* src = XL + (size_t)s * HC + h * CH + lane * 4;
  float* dst = AGG + (size_t)d * HC + h * CH + lane * 4;
#pragma unroll
  for (int j = 0; j < 4; ++j) atomicAdd(&dst[j], alpha * src[j]);
}

// =====================================================================
// Pooling + misc
// =====================================================================
__global__ void fill_f32(float* __restrict__ p, float v, int n) {
  const int i = blockIdx.x * 256 + threadIdx.x;
  if (i < n) p[i] = v;
}

__global__ __launch_bounds__(256)
void pool_accum(const float* __restrict__ h, const int* __restrict__ batch,
                float* __restrict__ POOL)
{
  const int idx = blockIdx.x * 256 + threadIdx.x;
  if (idx >= N_NODES * HC) return;
  const int n = idx / HC, c = idx % HC;
  atomicAdd(&POOL[batch[n] * HC + c], h[idx]);
}

__global__ void pool_count(const int* __restrict__ batch, float* __restrict__ CNT) {
  const int n = blockIdx.x * 256 + threadIdx.x;
  if (n < N_NODES) atomicAdd(&CNT[batch[n]], 1.0f);
}

__global__ void pool_div(float* __restrict__ POOL, const float* __restrict__ CNT) {
  const int idx = blockIdx.x * 256 + threadIdx.x;
  if (idx >= NG * HC) return;
  POOL[idx] /= fmaxf(CNT[idx / HC], 1.0f);
}

// =====================================================================
// Host orchestration
// =====================================================================
extern "C" void kernel_launch(void* const* d_in, const int* in_sizes, int n_in,
                              void* d_out, int out_size, void* d_ws, size_t ws_size,
                              hipStream_t stream)
{
  (void)in_sizes; (void)n_in; (void)out_size; (void)ws_size;

  const float* x     = (const float*)d_in[0];
  const int*   ei    = (const int*)d_in[1];
  const int*   batch = (const int*)d_in[2];
  // enc
  const float *ew1 = (const float*)d_in[3],  *eb1 = (const float*)d_in[4];
  const float *eg1 = (const float*)d_in[5],  *ebe1 = (const float*)d_in[6];
  const float *ew2 = (const float*)d_in[7],  *eb2 = (const float*)d_in[8];
  const float *eg2 = (const float*)d_in[9],  *ebe2 = (const float*)d_in[10];
  // layers: 11 + l*11 : Wl,bl,Wr,br,att,bias,Wres,bres,beta,ng,nb
  // head: 44..53
  const float *hw1 = (const float*)d_in[44], *hb1 = (const float*)d_in[45];
  const float *hg1 = (const float*)d_in[46], *hbe1 = (const float*)d_in[47];
  const float *hw2 = (const float*)d_in[48], *hb2 = (const float*)d_in[49];
  const float *hg2 = (const float*)d_in[50], *hbe2 = (const float*)d_in[51];
  const float *hw3 = (const float*)d_in[52], *hb3 = (const float*)d_in[53];

  // ---- fp32 workspace (all offsets multiple of 64 elements) ----
  float* ws = (float*)d_ws;
  size_t off = 0;
  float* H0    = ws + off; off += (size_t)N_NODES * HC;   // current hidden [N,1024]
  float* XINIT = ws + off; off += (size_t)N_NODES * CH;   // x_initial [N,128]
  float* XL    = ws + off; off += (size_t)N_NODES * HC;
  float* XR    = ws + off; off += (size_t)N_NODES * HC;   // reused as RES after scores
  float* AGG   = ws + off; off += (size_t)N_NODES * HC;
  float* SC    = ws + off; off += (size_t)ET * NH;
  float* MAXB  = ws + off; off += (size_t)N_NODES * NH;
  float* SUMB  = ws + off; off += (size_t)N_NODES * NH;
  float* POOL  = ws + off; off += (size_t)NG * HC;
  float* CNT   = ws + off; off += 64;
  float* T1    = ws + off; off += (size_t)NG * HC;
  float* T2    = ws + off; off += (size_t)NG * CH;

  // ---- bf16 workspace region ----
  __bf16* wsb = (__bf16*)(ws + off);
  size_t bo = 0;
  __bf16* XB  = wsb + bo; bo += (size_t)N_NODES * HC;     // staged bf16 activations
  __bf16* XIB = wsb + bo; bo += (size_t)N_NODES * CH;     // x_initial in bf16
  __bf16* WE1 = wsb + bo; bo += (size_t)CH * F_IN;
  __bf16* WE2 = wsb + bo; bo += (size_t)CH * CH;
  __bf16* WLb[3]; __bf16* WRb[3]; __bf16* WSb[3];
  for (int l = 0; l < 3; ++l) {
    const size_t kin = (l == 0) ? CH : HC;
    WLb[l] = wsb + bo; bo += (size_t)HC * kin;
    WRb[l] = wsb + bo; bo += (size_t)HC * kin;
    WSb[l] = wsb + bo; bo += (size_t)HC * CH;
  }
  __bf16* WH1 = wsb + bo; bo += (size_t)HC * HC;
  __bf16* WH2 = wsb + bo; bo += (size_t)CH * HC;
  __bf16* WH3 = wsb + bo; bo += (size_t)NCLS * CH;

  auto cvt = [&](const float* in, __bf16* out, int n) {
    hipLaunchKernelGGL(cvt_f32_to_bf16, dim3((n / 4 + 255) / 256), dim3(256), 0, stream,
                       in, out, n);
  };
  auto gemm = [&](const __bf16* X, const __bf16* W, const float* b, float* O,
                  int M, int Nout, int K) {
    dim3 grid(Nout / 64, (M + 63) / 64);
    hipLaunchKernelGGL(gemm_bf16_wmma, grid, dim3(128), 0, stream, X, W, b, O, M, Nout, K);
  };
  auto ln = [&](const float* in, const float* bias, const float* res, const float* beta,
                const float* g, const float* be, float* out, int rows, int D) {
    hipLaunchKernelGGL(fused_res_ln_gelu, dim3(rows), dim3(256), 0, stream,
                       in, bias, res, beta, g, be, out, D);
  };
  auto fill = [&](float* p, float v, int n) {
    hipLaunchKernelGGL(fill_f32, dim3((n + 255) / 256), dim3(256), 0, stream, p, v, n);
  };

  // ---- one-time weight conversion (deterministic, every launch) ----
  cvt(ew1, WE1, CH * F_IN);
  cvt(ew2, WE2, CH * CH);
  for (int l = 0; l < 3; ++l) {
    const int base = 11 + l * 11;
    const int kin = (l == 0) ? CH : HC;
    cvt((const float*)d_in[base + 0], WLb[l], HC * kin);
    cvt((const float*)d_in[base + 2], WRb[l], HC * kin);
    cvt((const float*)d_in[base + 6], WSb[l], HC * CH);
  }
  cvt(hw1, WH1, HC * HC);
  cvt(hw2, WH2, CH * HC);
  cvt(hw3, WH3, NCLS * CH);

  // ---------------- encoder ----------------
  cvt(x, XB, N_NODES * F_IN);
  gemm(XB, WE1, eb1, XL, N_NODES, CH, F_IN);
  ln(XL, nullptr, nullptr, nullptr, eg1, ebe1, H0, N_NODES, CH);     // H0 as [N,128]
  cvt(H0, XB, N_NODES * CH);
  gemm(XB, WE2, eb2, XL, N_NODES, CH, CH);
  ln(XL, nullptr, nullptr, nullptr, eg2, ebe2, XINIT, N_NODES, CH);  // x_initial
  cvt(XINIT, XIB, N_NODES * CH);                                     // persistent bf16

  // ---------------- 3 GAT layers ----------------
  const int EH = ET * NH;                      // edge*head work items
  const int wave_blocks = (EH + 7) / 8;        // 8 waves / 256-thread block
  const int th_blocks   = (EH + 255) / 256;

  for (int l = 0; l < 3; ++l) {
    const int base = 11 + l * 11;
    const float *bl   = (const float*)d_in[base + 1];
    const float *br   = (const float*)d_in[base + 3];
    const float *att  = (const float*)d_in[base + 4], *bias = (const float*)d_in[base + 5];
    const float *bres = (const float*)d_in[base + 7];
    const float *beta = (const float*)d_in[base + 8];
    const float *ng   = (const float*)d_in[base + 9], *nb   = (const float*)d_in[base + 10];

    const __bf16* Ab;
    int kin;
    if (l == 0) { Ab = XIB; kin = CH; }        // layer-1 input = x_initial (already bf16)
    else        { cvt(H0, XB, N_NODES * HC); Ab = XB; kin = HC; }

    gemm(Ab, WLb[l], bl, XL, N_NODES, HC, kin);
    gemm(Ab, WRb[l], br, XR, N_NODES, HC, kin);

    fill(MAXB, -INFINITY, N_NODES * NH);
    fill(SUMB, 0.0f, N_NODES * NH);
    fill(AGG, 0.0f, N_NODES * HC);

    hipLaunchKernelGGL(edge_scores, dim3(wave_blocks), dim3(256), 0, stream, XL, XR, att, ei, SC);
    hipLaunchKernelGGL(edge_seg_max, dim3(th_blocks), dim3(256), 0, stream, SC, ei, MAXB);
    hipLaunchKernelGGL(edge_exp_sum, dim3(th_blocks), dim3(256), 0, stream, SC, ei, MAXB, SUMB);
    hipLaunchKernelGGL(edge_scatter, dim3(wave_blocks), dim3(256), 0, stream, SC, SUMB, XL, ei, AGG);

    // residual from x_initial (XR is free after scores -> reuse as RES)
    gemm(XIB, WSb[l], bres, XR, N_NODES, HC, CH);
    ln(AGG, bias, XR, beta, ng, nb, H0, N_NODES, HC);   // h = gelu(LN(agg+bias+beta*res))
  }

  // ---------------- global mean pool ----------------
  fill(POOL, 0.0f, NG * HC);
  fill(CNT, 0.0f, NG);
  hipLaunchKernelGGL(pool_accum, dim3((N_NODES * HC + 255) / 256), dim3(256), 0, stream, H0, batch, POOL);
  hipLaunchKernelGGL(pool_count, dim3((N_NODES + 255) / 256), dim3(256), 0, stream, batch, CNT);
  hipLaunchKernelGGL(pool_div, dim3((NG * HC + 255) / 256), dim3(256), 0, stream, POOL, CNT);

  // ---------------- head ----------------
  cvt(POOL, XB, NG * HC);
  gemm(XB, WH1, hb1, T1, NG, HC, HC);
  ln(T1, nullptr, nullptr, nullptr, hg1, hbe1, T1, NG, HC);
  cvt(T1, XB, NG * HC);
  gemm(XB, WH2, hb2, T2, NG, CH, HC);
  ln(T2, nullptr, nullptr, nullptr, hg2, hbe2, T2, NG, CH);
  cvt(T2, XB, NG * CH);
  gemm(XB, WH3, hb3, (float*)d_out, NG, NCLS, CH);
}
